// LSTM_decoder_31018253812111
// MI455X (gfx1250) — matrix operand order repaired and checked by reading, not compiled, not run on hardware
//
#include <hip/hip_runtime.h>
#include <cstdint>
#include <cstddef>

typedef __bf16 bf16;
typedef __attribute__((ext_vector_type(16))) __bf16 v16bf;
typedef __attribute__((ext_vector_type(8)))  float  v8f;

// ---------------- problem constants ----------------
constexpr int T_STEPS = 512;
constexpr int IN_D    = 256;
constexpr int B       = 128;
constexpr int H       = 512;
constexpr int TAG     = 25;
constexpr int LM      = 200;
constexpr int PB      = 6;
constexpr int CB      = 2;
constexpr int G4H     = 2048;   // 4*H
constexpr int G3H     = 1536;   // 3*H
constexpr int KX      = 480;    // 256+200+6+2 = 464, padded to 480 (15 k-tiles of 32)
constexpr int KAUX    = 224;    // 208 padded to 224 (7 k-tiles)
constexpr int NWG     = 20;     // persistent scan workgroups
constexpr int TAGP    = 32;     // TAG padded to 2 M-tiles

// ---------------- workspace layout (bytes) ----------------
constexpr size_t OFF_BAR   = 0;
constexpr size_t OFF_WCOMB = 256;
constexpr size_t OFF_WTAUX = OFF_WCOMB + (size_t)G4H * KX * 2;
constexpr size_t OFF_WGATE = OFF_WTAUX + (size_t)H * KAUX * 2;
constexpr size_t OFF_WTI   = OFF_WGATE + (size_t)G4H * H * 2;
constexpr size_t OFF_WCO   = OFF_WTI   + (size_t)G3H * H * 2;
constexpr size_t OFF_WHT   = OFF_WCO   + (size_t)H * H * 2;
constexpr size_t OFF_X     = OFF_WHT   + (size_t)H * H * 2;
constexpr size_t OFF_II    = OFF_X     + (size_t)T_STEPS * B * KX * 2;
constexpr size_t OFF_TAUX  = OFF_II    + (size_t)T_STEPS * G4H * B * 4;
constexpr size_t OFF_TS    = OFF_TAUX  + (size_t)T_STEPS * H * B * 4;
constexpr size_t OFF_GATES = OFF_TS    + (size_t)T_STEPS * H * B * 4;
constexpr size_t OFF_C     = OFF_GATES + (size_t)G4H * B * 4;
constexpr size_t OFF_HBF   = OFF_C     + (size_t)H * B * 4;
constexpr size_t OFF_TBF   = OFF_HBF   + (size_t)H * B * 2;
constexpr size_t OFF_CBF   = OFF_TBF   + (size_t)H * B * 2;
constexpr size_t OFF_YF    = OFF_CBF   + (size_t)H * B * 2;
constexpr size_t OFF_YC    = OFF_YF    + (size_t)T_STEPS * TAG * B * 4;
constexpr size_t OFF_WYF   = OFF_YC    + (size_t)T_STEPS * TAG * B * 4;
constexpr size_t OFF_WYC   = OFF_WYF   + (size_t)TAGP * H * 2;
constexpr size_t OFF_TSBF  = OFF_WYC   + (size_t)TAGP * H * 2;  // T*B*H bf16, [t][b][h]

// ---------------- WMMA fragment helpers (wave32, 16x16x32 bf16) ----------------
// A: 16x32 (MxK) from row-major W[m][k]; lane L: M=L&15, g=L>>4,
// elements 0..7 -> K = g*8+{0..7}, elements 8..15 -> K = 16+g*8+{0..7}.
__device__ __forceinline__ v16bf load_a_frag(const bf16* W, int ld, int lane) {
  const int m = lane & 15;
  const int g = lane >> 4;
  const bf16* p = W + (size_t)m * ld + g * 8;
  v16bf a;
#pragma unroll
  for (int e = 0; e < 8; ++e) a[e] = p[e];
#pragma unroll
  for (int e = 0; e < 8; ++e) a[8 + e] = p[16 + e];
  return a;
}

// B: 32x16 (KxN) from activations stored [n][k]; lane L: N=L&15, g=L>>4,
// elements 0..7 -> K = g*8+{0..7}, elements 8..15 -> K = 16+g*8+{0..7}.
__device__ __forceinline__ v16bf load_b_frag(const bf16* Act, int ld, int lane) {
  const int n = lane & 15;
  const int g = lane >> 4;
  const bf16* p = Act + (size_t)n * ld + g * 8;
  v16bf b;
#pragma unroll
  for (int e = 0; e < 8; ++e) b[e] = p[e];
#pragma unroll
  for (int e = 0; e < 8; ++e) b[8 + e] = p[16 + e];
  return b;
}

// C/D 16x16 f32: VGPR r -> M = r + 8*(lane>>4), N = lane&15
__device__ __forceinline__ void store_c(float* C, int ldc, int lane, const v8f acc) {
  const int n = lane & 15;
  const int g = lane >> 4;
#pragma unroll
  for (int r = 0; r < 8; ++r) C[(size_t)(g * 8 + r) * ldc + n] = acc[r];
}

__device__ __forceinline__ v8f zero8() {
  v8f z;
#pragma unroll
  for (int i = 0; i < 8; ++i) z[i] = 0.0f;
  return z;
}

// One wave computes a 32(M) x 64(N) f32 tile (2 M-tiles x 4 N-tiles), with
// register double-buffering: fragments for k-chunk kc+1 are loaded while the
// 8 WMMAs of chunk kc issue, so VMEM latency overlaps the matrix pipe.
template <int KT>
__device__ __forceinline__ void wave_gemm2x4(const bf16* __restrict__ W, int ldw,
                                             const bf16* __restrict__ Act, int lda,
                                             v8f acc[2][4], int lane) {
  v16bf a0[2], b0[4], a1[2], b1[4];
#pragma unroll
  for (int i = 0; i < 2; ++i) a0[i] = load_a_frag(W + (size_t)(i * 16) * ldw, ldw, lane);
#pragma unroll
  for (int j = 0; j < 4; ++j) b0[j] = load_b_frag(Act + (size_t)(j * 16) * lda, lda, lane);
#pragma unroll
  for (int kc = 0; kc < KT; ++kc) {
    if (kc + 1 < KT) {
      const bf16* Wn = W + (kc + 1) * 32;
      const bf16* An = Act + (kc + 1) * 32;
#pragma unroll
      for (int i = 0; i < 2; ++i) a1[i] = load_a_frag(Wn + (size_t)(i * 16) * ldw, ldw, lane);
#pragma unroll
      for (int j = 0; j < 4; ++j) b1[j] = load_b_frag(An + (size_t)(j * 16) * lda, lda, lane);
    }
#pragma unroll
    for (int i = 0; i < 2; ++i)
#pragma unroll
      for (int j = 0; j < 4; ++j)
        acc[i][j] = __builtin_amdgcn_wmma_f32_16x16x32_bf16(
            false, a0[i], false, b0[j], (short)0, acc[i][j], false, false);
    if (kc + 1 < KT) {
#pragma unroll
      for (int i = 0; i < 2; ++i) a0[i] = a1[i];
#pragma unroll
      for (int j = 0; j < 4; ++j) b0[j] = b1[j];
    }
  }
}

// ---------------- device-scope grid barrier ----------------
__device__ __forceinline__ void grid_sync(unsigned* bar) {
  __threadfence();
  __syncthreads();
  if (threadIdx.x == 0) {
    unsigned* cnt = bar;
    unsigned* gen = bar + 1;
    unsigned g = __hip_atomic_load(gen, __ATOMIC_RELAXED, __HIP_MEMORY_SCOPE_AGENT);
    unsigned old = __hip_atomic_fetch_add(cnt, 1u, __ATOMIC_ACQ_REL, __HIP_MEMORY_SCOPE_AGENT);
    if (old == (unsigned)(NWG - 1)) {
      __hip_atomic_store(cnt, 0u, __ATOMIC_RELAXED, __HIP_MEMORY_SCOPE_AGENT);
      __hip_atomic_fetch_add(gen, 1u, __ATOMIC_RELEASE, __HIP_MEMORY_SCOPE_AGENT);
    } else {
      while (__hip_atomic_load(gen, __ATOMIC_ACQUIRE, __HIP_MEMORY_SCOPE_AGENT) == g) {
        __builtin_amdgcn_s_sleep(2);
      }
    }
  }
  __syncthreads();
}

// ---------------- setup kernels ----------------
__global__ void k_init(const float* __restrict__ h0, const float* __restrict__ c0,
                       const float* __restrict__ t0, float* __restrict__ cstate,
                       bf16* __restrict__ hbf, bf16* __restrict__ tbf, unsigned* bar) {
  int idx = blockIdx.x * blockDim.x + threadIdx.x;
  if (idx == 0) { bar[0] = 0u; bar[1] = 0u; }
  if (idx >= H * B) return;
  int h = idx >> 7, b = idx & (B - 1);
  cstate[idx] = c0[idx];
  hbf[(size_t)b * H + h] = (bf16)h0[idx];
  tbf[(size_t)b * H + h] = (bf16)t0[idx];
}

// Wcomb = [w_ii | w_ii*w_lmw^T | w_ii*w_posw^T | w_ii*w_capw^T | 0-pad]  (G4H x KX)
__global__ void k_pack_wcomb(const float* __restrict__ w_ii, const float* __restrict__ w_lmw,
                             const float* __restrict__ w_posw, const float* __restrict__ w_capw,
                             bf16* __restrict__ Wcomb) {
  int idx = blockIdx.x * blockDim.x + threadIdx.x;
  if (idx >= G4H * KX) return;
  int g = idx / KX, k = idx - g * KX;
  float v = 0.0f;
  if (k < IN_D) {
    v = w_ii[(size_t)g * IN_D + k];
  } else if (k < IN_D + LM) {
    const float* wr = w_lmw + (size_t)(k - IN_D) * IN_D;
    const float* wi = w_ii + (size_t)g * IN_D;
    float a = 0.0f;
    for (int i = 0; i < IN_D; ++i) a += wi[i] * wr[i];
    v = a;
  } else if (k < IN_D + LM + PB) {
    const float* wr = w_posw + (size_t)(k - IN_D - LM) * IN_D;
    const float* wi = w_ii + (size_t)g * IN_D;
    float a = 0.0f;
    for (int i = 0; i < IN_D; ++i) a += wi[i] * wr[i];
    v = a;
  } else if (k < IN_D + LM + PB + CB) {
    const float* wr = w_capw + (size_t)(k - IN_D - LM - PB) * IN_D;
    const float* wi = w_ii + (size_t)g * IN_D;
    float a = 0.0f;
    for (int i = 0; i < IN_D; ++i) a += wi[i] * wr[i];
    v = a;
  }
  Wcomb[idx] = (bf16)v;
}

__global__ void k_pack_bf16(const float* __restrict__ src, bf16* __restrict__ dst, int n) {
  int idx = blockIdx.x * blockDim.x + threadIdx.x;
  if (idx < n) dst[idx] = (bf16)src[idx];
}

__global__ void k_pack_wtaux(const float* __restrict__ w_lmt, const float* __restrict__ w_post,
                             const float* __restrict__ w_capt, bf16* __restrict__ Wtaux) {
  int idx = blockIdx.x * blockDim.x + threadIdx.x;
  if (idx >= H * KAUX) return;
  int h = idx / KAUX, k = idx - h * KAUX;
  float v = 0.0f;
  if (k < LM)                v = w_lmt[(size_t)h * LM + k];
  else if (k < LM + PB)      v = w_post[(size_t)h * PB + (k - LM)];
  else if (k < LM + PB + CB) v = w_capt[(size_t)h * CB + (k - LM - PB)];
  Wtaux[idx] = (bf16)v;
}

// Wyf/Wyc: TAG rows padded to 32, H cols, bf16
__global__ void k_pack_wy(const float* __restrict__ wf, const float* __restrict__ wc,
                          bf16* __restrict__ Wyf, bf16* __restrict__ Wyc) {
  int idx = blockIdx.x * blockDim.x + threadIdx.x;
  if (idx >= TAGP * H) return;
  int g = idx / H, k = idx - g * H;
  float vf = 0.0f, vc = 0.0f;
  if (g < TAG) {
    vf = wf[(size_t)g * H + k];
    vc = wc[(size_t)g * H + k];
  }
  Wyf[idx] = (bf16)vf;
  Wyc[idx] = (bf16)vc;
}

// X[t][b][k] bf16 : concat(inputs, lms, pos, caps) along k, zero pad to KX
__global__ void k_packX(const float* __restrict__ inputs, const float* __restrict__ lms,
                        const float* __restrict__ pos, const float* __restrict__ caps,
                        bf16* __restrict__ X) {
  int idx = blockIdx.x * blockDim.x + threadIdx.x;
  if (idx >= T_STEPS * B * KX) return;
  int t = idx / (B * KX);
  int r = idx - t * (B * KX);
  int b = r / KX;
  int k = r - b * KX;
  float v = 0.0f;
  if (k < IN_D)                     v = inputs[((size_t)t * IN_D + k) * B + b];
  else if (k < IN_D + LM)           v = lms[((size_t)t * LM + (k - IN_D)) * B + b];
  else if (k < IN_D + LM + PB)      v = pos[((size_t)t * PB + (k - IN_D - LM)) * B + b];
  else if (k < IN_D + LM + PB + CB) v = caps[((size_t)t * CB + (k - IN_D - LM - PB)) * B + b];
  X[idx] = (bf16)v;
}

// ---------------- big parallel GEMMs ----------------
__global__ __launch_bounds__(256) void k_gemm_ii(const bf16* __restrict__ Wcomb,
                                                 const bf16* __restrict__ X,
                                                 float* __restrict__ ii_all) {
  const int t = blockIdx.x;
  const int mt = blockIdx.y;
  const int wave = threadIdx.x >> 5, lane = threadIdx.x & 31;
  const int m0 = mt * 128 + (wave >> 1) * 32;
  const int n0 = (wave & 1) * 64;
  const bf16* Xt = X + (size_t)t * B * KX + (size_t)n0 * KX;
  v8f acc[2][4];
#pragma unroll
  for (int i = 0; i < 2; ++i)
#pragma unroll
    for (int j = 0; j < 4; ++j) acc[i][j] = zero8();
  wave_gemm2x4<KX / 32>(Wcomb + (size_t)m0 * KX, KX, Xt, KX, acc, lane);
  float* out = ii_all + (size_t)t * G4H * B;
#pragma unroll
  for (int i = 0; i < 2; ++i)
#pragma unroll
    for (int j = 0; j < 4; ++j)
      store_c(out + (size_t)(m0 + i * 16) * B + n0 + j * 16, B, lane, acc[i][j]);
}

__global__ __launch_bounds__(256) void k_gemm_taux(const bf16* __restrict__ Wtaux,
                                                   const bf16* __restrict__ X,
                                                   float* __restrict__ taux) {
  const int t = blockIdx.x;
  const int mt = blockIdx.y;
  const int wave = threadIdx.x >> 5, lane = threadIdx.x & 31;
  const int m0 = mt * 128 + (wave >> 1) * 32;
  const int n0 = (wave & 1) * 64;
  const bf16* Xt = X + (size_t)t * B * KX + (size_t)n0 * KX + IN_D;  // k = 256..479
  v8f acc[2][4];
#pragma unroll
  for (int i = 0; i < 2; ++i)
#pragma unroll
    for (int j = 0; j < 4; ++j) acc[i][j] = zero8();
  wave_gemm2x4<KAUX / 32>(Wtaux + (size_t)m0 * KAUX, KAUX, Xt, KX, acc, lane);
  float* out = taux + (size_t)t * H * B;
#pragma unroll
  for (int i = 0; i < 2; ++i)
#pragma unroll
    for (int j = 0; j < 4; ++j)
      store_c(out + (size_t)(m0 + i * 16) * B + n0 + j * 16, B, lane, acc[i][j]);
}

// ---------------- persistent cooperative LSTM scan ----------------
__global__ __launch_bounds__(256) void k_scan(
    const float* __restrict__ ii_all, const float* __restrict__ taux,
    const float* __restrict__ b_i,
    const bf16* __restrict__ Wgate, const bf16* __restrict__ Wti,
    const bf16* __restrict__ Wco, const bf16* __restrict__ Wht,
    float* __restrict__ gates, float* __restrict__ cstate,
    bf16* __restrict__ hbf, bf16* __restrict__ tbf, bf16* __restrict__ cbf,
    float* __restrict__ Ts, bf16* __restrict__ Tsbf, unsigned* bar) {
  const int wg = blockIdx.x;
  const int tid = threadIdx.x;
  const int wave = tid >> 5;
  const int lane = tid & 31;
  const int n = lane & 15;
  const int gh = lane >> 4;
  const int mw = (wave >> 1) * 32;
  const int n0 = (wave & 1) * 64;

  for (int t = 0; t < T_STEPS; ++t) {
    const float* ii_t = ii_all + (size_t)t * G4H * B;

    // P0: gates[m][b] = Wgate@h (+ Wti@t for m < 3H)   (WGs 0..15, 128 rows each)
    if (wg < 16) {
      const int m0 = wg * 128 + mw;
      v8f acc[2][4];
#pragma unroll
      for (int i = 0; i < 2; ++i)
#pragma unroll
        for (int j = 0; j < 4; ++j) acc[i][j] = zero8();
      wave_gemm2x4<H / 32>(Wgate + (size_t)m0 * H, H, hbf + (size_t)n0 * H, H, acc, lane);
      if (m0 < G3H)
        wave_gemm2x4<H / 32>(Wti + (size_t)m0 * H, H, tbf + (size_t)n0 * H, H, acc, lane);
#pragma unroll
      for (int i = 0; i < 2; ++i)
#pragma unroll
        for (int j = 0; j < 4; ++j)
          store_c(gates + (size_t)(m0 + i * 16) * B + n0 + j * 16, B, lane, acc[i][j]);
    }
    grid_sync(bar);

    // P1: cell update (all WGs)
    for (int idx = wg * 256 + tid; idx < H * B; idx += NWG * 256) {
      const int h = idx >> 7;
      const int b = idx & (B - 1);
      const float pi = gates[(size_t)h * B + b]           + ii_t[(size_t)h * B + b]           + b_i[h];
      const float pf = gates[(size_t)(H + h) * B + b]     + ii_t[(size_t)(H + h) * B + b]     + b_i[H + h];
      const float pz = gates[(size_t)(2 * H + h) * B + b] + ii_t[(size_t)(2 * H + h) * B + b] + b_i[2 * H + h];
      const float ig = 1.0f / (1.0f + expf(-pi));
      const float fg = 1.0f / (1.0f + expf(-pf));
      const float zg = tanhf(pz);
      const float cn = fg * cstate[idx] + ig * zg;
      cstate[idx] = cn;
      cbf[(size_t)b * H + h] = (bf16)cn;
    }
    grid_sync(bar);

    // P2: o = sigmoid(ii4 + hi4 + Wco@c_new + b4); h_new = o*tanh(c_new)   (WGs 16..19)
    if (wg >= 16) {
      const int m0 = (wg - 16) * 128 + mw;
      v8f acc[2][4];
#pragma unroll
      for (int i = 0; i < 2; ++i)
#pragma unroll
        for (int j = 0; j < 4; ++j) acc[i][j] = zero8();
      wave_gemm2x4<H / 32>(Wco + (size_t)m0 * H, H, cbf + (size_t)n0 * H, H, acc, lane);
#pragma unroll
      for (int i = 0; i < 2; ++i) {
#pragma unroll
        for (int j = 0; j < 4; ++j) {
#pragma unroll
          for (int r = 0; r < 8; ++r) {
            const int m = m0 + i * 16 + gh * 8 + r;
            const int b = n0 + j * 16 + n;
            float og = acc[i][j][r] + gates[(size_t)(G3H + m) * B + b]
                     + ii_t[(size_t)(G3H + m) * B + b] + b_i[G3H + m];
            og = 1.0f / (1.0f + expf(-og));
            const float hn = og * tanhf(cstate[(size_t)m * B + b]);
            hbf[(size_t)b * H + m] = (bf16)hn;
          }
        }
      }
    }
    grid_sync(bar);

    // P3: t_new = Wht@h_new + b5 + taux    (WGs 16..19)
    if (wg >= 16) {
      const int m0 = (wg - 16) * 128 + mw;
      v8f acc[2][4];
#pragma unroll
      for (int i = 0; i < 2; ++i)
#pragma unroll
        for (int j = 0; j < 4; ++j) acc[i][j] = zero8();
      wave_gemm2x4<H / 32>(Wht + (size_t)m0 * H, H, hbf + (size_t)n0 * H, H, acc, lane);
      const float* taux_t = taux + (size_t)t * H * B;
      float* Ts_t = Ts + (size_t)t * H * B;
      bf16* Tsbf_t = Tsbf + (size_t)t * B * H;
#pragma unroll
      for (int i = 0; i < 2; ++i) {
#pragma unroll
        for (int j = 0; j < 4; ++j) {
#pragma unroll
          for (int r = 0; r < 8; ++r) {
            const int m = m0 + i * 16 + gh * 8 + r;
            const int b = n0 + j * 16 + n;
            const float tn = acc[i][j][r] + b_i[4 * H + m] + taux_t[(size_t)m * B + b];
            Ts_t[(size_t)m * B + b] = tn;
            tbf[(size_t)b * H + m] = (bf16)tn;
            Tsbf_t[(size_t)b * H + m] = (bf16)tn;
          }
        }
      }
    }
    grid_sync(bar);
  }
}

// ---------------- output head (WMMA: M=32 padded from TAG, K=H, N=B) ----------------
__global__ __launch_bounds__(256) void k_y2(const bf16* __restrict__ Wyf,
                                            const bf16* __restrict__ Wyc,
                                            const bf16* __restrict__ Tsbf,
                                            const float* __restrict__ bfv,
                                            const float* __restrict__ bcv,
                                            float* __restrict__ yf, float* __restrict__ yc) {
  const int t = blockIdx.x;
  const int wave = threadIdx.x >> 5, lane = threadIdx.x & 31;
  const int n0 = wave * 16;             // 8 waves x 16 cols = 128
  const int n = lane & 15, gh = lane >> 4;
  const bf16* Act = Tsbf + (size_t)t * B * H + (size_t)n0 * H;  // [n][k], lda=H
  v8f accf[2], accc[2];
#pragma unroll
  for (int i = 0; i < 2; ++i) { accf[i] = zero8(); accc[i] = zero8(); }
  for (int kc = 0; kc < H / 32; ++kc) {
    v16bf b = load_b_frag(Act + kc * 32, H, lane);
#pragma unroll
    for (int i = 0; i < 2; ++i) {
      v16bf af = load_a_frag(Wyf + (size_t)(i * 16) * H + kc * 32, H, lane);
      v16bf ac = load_a_frag(Wyc + (size_t)(i * 16) * H + kc * 32, H, lane);
      accf[i] = __builtin_amdgcn_wmma_f32_16x16x32_bf16(false, af, false, b, (short)0, accf[i], false, false);
      accc[i] = __builtin_amdgcn_wmma_f32_16x16x32_bf16(false, ac, false, b, (short)0, accc[i], false, false);
    }
  }
#pragma unroll
  for (int i = 0; i < 2; ++i) {
#pragma unroll
    for (int r = 0; r < 8; ++r) {
      const int g = i * 16 + gh * 8 + r;
      if (g < TAG) {
        const int bcol = n0 + n;
        yf[(size_t)t * TAG * B + (size_t)g * B + bcol] = accf[i][r] + bfv[g];
        yc[(size_t)t * TAG * B + (size_t)g * B + bcol] = accc[i][r] + bcv[g];
      }
    }
  }
}

__global__ void k_out(const float* __restrict__ yf, const float* __restrict__ yc,
                      float* __restrict__ of, float* __restrict__ oc,
                      float* __restrict__ odf, float* __restrict__ odc) {
  int idx = blockIdx.x * blockDim.x + threadIdx.x;
  if (idx >= T_STEPS * B) return;
  int t = idx / B, b = idx - t * B;
  const float* f = yf + (size_t)t * TAG * B + b;
  const float* c = yc + (size_t)t * TAG * B + b;
  float mf = -1e30f, mc = -1e30f;
  for (int g = 0; g < TAG; ++g) {
    mf = fmaxf(mf, f[(size_t)g * B]);
    mc = fmaxf(mc, c[(size_t)g * B]);
  }
  float sf = 0.0f, sc = 0.0f;
  for (int g = 0; g < TAG; ++g) {
    sf += expf(f[(size_t)g * B] - mf);
    sc += expf(c[(size_t)g * B] - mc);
  }
  const float lf = mf + logf(sf);
  const float lc = mc + logf(sc);
  size_t ob = ((size_t)b * T_STEPS + t) * TAG;
  for (int g = 0; g < TAG; ++g) {
    const float vf = f[(size_t)g * B];
    const float vc = c[(size_t)g * B];
    of[ob + g] = vf - lf;
    oc[ob + g] = vc - lc;
    odf[ob + g] = vf;
    odc[ob + g] = vc;
  }
}

__global__ void k_hidden(const float* __restrict__ Ts, float* __restrict__ hid) {
  int idx = blockIdx.x * blockDim.x + threadIdx.x;
  if (idx >= B * T_STEPS * H) return;
  int b = idx / (T_STEPS * H);
  int r = idx - b * (T_STEPS * H);
  int t = r / H;
  int h = r - t * H;
  hid[idx] = Ts[((size_t)t * H + h) * B + b];
}

// ---------------- host launch ----------------
extern "C" void kernel_launch(void* const* d_in, const int* in_sizes, int n_in,
                              void* d_out, int out_size, void* d_ws, size_t ws_size,
                              hipStream_t stream) {
  const float* inputs   = (const float*)d_in[0];
  const float* lms      = (const float*)d_in[1];
  const float* poses    = (const float*)d_in[2];
  const float* caps     = (const float*)d_in[3];
  const float* h0       = (const float*)d_in[4];
  const float* c0       = (const float*)d_in[5];
  const float* t0       = (const float*)d_in[6];
  const float* w_ii     = (const float*)d_in[7];
  const float* w_hi     = (const float*)d_in[8];
  const float* w_ti     = (const float*)d_in[9];
  const float* w_co     = (const float*)d_in[10];
  const float* w_ht     = (const float*)d_in[11];
  const float* b_i      = (const float*)d_in[12];
  const float* w_y_fact = (const float*)d_in[13];
  const float* b_y_fact = (const float*)d_in[14];
  const float* w_y_cond = (const float*)d_in[15];
  const float* b_y_cond = (const float*)d_in[16];
  const float* w_lmw    = (const float*)d_in[17];
  const float* w_posw   = (const float*)d_in[18];
  const float* w_capw   = (const float*)d_in[19];
  const float* w_lmt    = (const float*)d_in[20];
  const float* w_post   = (const float*)d_in[21];
  const float* w_capt   = (const float*)d_in[22];

  char* ws = (char*)d_ws;
  unsigned* bar = (unsigned*)(ws + OFF_BAR);
  bf16* Wcomb = (bf16*)(ws + OFF_WCOMB);
  bf16* Wtaux = (bf16*)(ws + OFF_WTAUX);
  bf16* Wgate = (bf16*)(ws + OFF_WGATE);
  bf16* Wti   = (bf16*)(ws + OFF_WTI);
  bf16* Wco   = (bf16*)(ws + OFF_WCO);
  bf16* Wht   = (bf16*)(ws + OFF_WHT);
  bf16* X     = (bf16*)(ws + OFF_X);
  float* ii_all = (float*)(ws + OFF_II);
  float* tauxp  = (float*)(ws + OFF_TAUX);
  float* Tsp    = (float*)(ws + OFF_TS);
  float* gates  = (float*)(ws + OFF_GATES);
  float* cstate = (float*)(ws + OFF_C);
  bf16* hbf  = (bf16*)(ws + OFF_HBF);
  bf16* tbf  = (bf16*)(ws + OFF_TBF);
  bf16* cbf  = (bf16*)(ws + OFF_CBF);
  float* yf  = (float*)(ws + OFF_YF);
  float* yc  = (float*)(ws + OFF_YC);
  bf16* Wyf  = (bf16*)(ws + OFF_WYF);
  bf16* Wyc  = (bf16*)(ws + OFF_WYC);
  bf16* Tsbf = (bf16*)(ws + OFF_TSBF);

  const size_t N_OF = (size_t)B * T_STEPS * TAG;
  float* of  = (float*)d_out;
  float* oc  = of + N_OF;
  float* odf = of + 2 * N_OF;
  float* odc = of + 3 * N_OF;
  float* hid = of + 4 * N_OF;

  auto cdiv = [](long long a, long long b) { return (unsigned)((a + b - 1) / b); };

  k_init<<<cdiv(H * B, 256), 256, 0, stream>>>(h0, c0, t0, cstate, hbf, tbf, bar);
  k_pack_wcomb<<<cdiv((long long)G4H * KX, 256), 256, 0, stream>>>(w_ii, w_lmw, w_posw, w_capw, Wcomb);
  k_pack_bf16<<<cdiv((long long)G4H * H, 256), 256, 0, stream>>>(w_hi, Wgate, G4H * H);
  k_pack_bf16<<<cdiv((long long)G3H * H, 256), 256, 0, stream>>>(w_ti, Wti, G3H * H);
  k_pack_bf16<<<cdiv((long long)H * H, 256), 256, 0, stream>>>(w_co, Wco, H * H);
  k_pack_bf16<<<cdiv((long long)H * H, 256), 256, 0, stream>>>(w_ht, Wht, H * H);
  k_pack_wtaux<<<cdiv((long long)H * KAUX, 256), 256, 0, stream>>>(w_lmt, w_post, w_capt, Wtaux);
  k_pack_wy<<<cdiv((long long)TAGP * H, 256), 256, 0, stream>>>(w_y_fact, w_y_cond, Wyf, Wyc);
  k_packX<<<cdiv((long long)T_STEPS * B * KX, 256), 256, 0, stream>>>(inputs, lms, poses, caps, X);
  k_gemm_ii<<<dim3(T_STEPS, G4H / 128), 256, 0, stream>>>(Wcomb, X, ii_all);
  k_gemm_taux<<<dim3(T_STEPS, H / 128), 256, 0, stream>>>(Wtaux, X, tauxp);
  k_scan<<<NWG, 256, 0, stream>>>(ii_all, tauxp, b_i, Wgate, Wti, Wco, Wht,
                                  gates, cstate, hbf, tbf, cbf, Tsp, Tsbf, bar);
  k_y2<<<T_STEPS, 256, 0, stream>>>(Wyf, Wyc, Tsbf, b_y_fact, b_y_cond, yf, yc);
  k_out<<<cdiv((long long)T_STEPS * B, 256), 256, 0, stream>>>(yf, yc, of, oc, odf, odc);
  k_hidden<<<cdiv((long long)B * T_STEPS * H, 256), 256, 0, stream>>>(Tsp, hid);
}
